// Ami_AttGNN_33337536152075
// MI455X (gfx1250) — matrix operand order repaired and checked by reading, not compiled, not run on hardware
//
#include <hip/hip_runtime.h>
#include <hip/hip_bf16.h>

typedef __attribute__((ext_vector_type(16))) __bf16        v16bf;
typedef __attribute__((ext_vector_type(8)))  float         v8f;
typedef __attribute__((ext_vector_type(4)))  unsigned int  v4u;

union ABu { unsigned int u[8]; v4u q[2]; v16bf v; };

__device__ __forceinline__ unsigned short bf16_bits(__bf16 b) {
    union { __bf16 b; unsigned short u; } cv; cv.b = b; return cv.u;
}
__device__ __forceinline__ float lrelu(float v, float s) { return v > 0.f ? v : s * v; }

// order-preserving float -> u32 key (monotone, supports -inf)
__device__ __forceinline__ unsigned fkey(float f) {
    unsigned b = __float_as_uint(f);
    return (b & 0x80000000u) ? ~b : (b | 0x80000000u);
}
__device__ __forceinline__ float funkey(unsigned k) {
    unsigned b = (k & 0x80000000u) ? (k & 0x7fffffffu) : ~k;
    return __uint_as_float(b);
}
#define KEY_NEG_INF 0x007FFFFFu

// ---------------------------------------------------------------------------
// Pack W (512x512 fp32, row-major [K][Ncol]) into WMMA B-layout, hi/lo bf16.
// B tile (ntile,kstep): lanes 0-15 -> n=lane, K=2v,2v+1 (v<4), K=8+2v (v>=4) ; lanes 16-31 -> K+8.
// Flat dword index: ((ntile*16 + ks)*32 + lane)*8 + v
// ---------------------------------------------------------------------------
__global__ void pack_w_kernel(const float* __restrict__ W,
                              unsigned int* __restrict__ bhi,
                              unsigned int* __restrict__ blo) {
    int t = blockIdx.x * blockDim.x + threadIdx.x;
    if (t >= 32 * 16 * 32 * 8) return;
    int v     = t & 7;
    int lane  = (t >> 3) & 31;
    int ks    = (t >> 8) & 15;
    int ntile = t >> 12;
    int nn  = lane & 15;
    int kk0 = ((lane < 16) ? 0 : 8) + 2 * v + ((v < 4) ? 0 : 8);
    int k = ks * 32 + kk0;
    int n = ntile * 16 + nn;
    float w0 = W[k * 512 + n];
    float w1 = W[(k + 1) * 512 + n];
    __bf16 h0 = (__bf16)w0; __bf16 l0 = (__bf16)(w0 - (float)h0);
    __bf16 h1 = (__bf16)w1; __bf16 l1 = (__bf16)(w1 - (float)h1);
    bhi[t] = (unsigned)bf16_bits(h0) | ((unsigned)bf16_bits(h1) << 16);
    blo[t] = (unsigned)bf16_bits(l0) | ((unsigned)bf16_bits(l1) << 16);
}

// x (fp32 row-major) -> hi/lo bf16 row-major copies
__global__ void conv_x_kernel(const float* __restrict__ x,
                              unsigned short* __restrict__ xhi,
                              unsigned short* __restrict__ xlo, int total) {
    int t = blockIdx.x * blockDim.x + threadIdx.x;
    if (t >= total) return;
    float v = x[t];
    __bf16 hb = (__bf16)v;
    __bf16 lb = (__bf16)(v - (float)hb);
    xhi[t] = bf16_bits(hb);
    xlo[t] = bf16_bits(lb);
}

// ---------------------------------------------------------------------------
// h = x @ W via split-bf16 WMMA: c += Ah*Bh + Ah*Bl + Al*Bh
// One wave computes a 16x64 strip: 4 n-tiles sharing one A fetch per k-step.
// 4 independent accumulators break the WMMA RAW chain (XDL latency hiding).
// All operand fetches are explicit b128 (uint4) loads.
// A layout (16-bit, 16x32): lane m=lane&15, half=lane>>4;
//   dwords v0..3 at byte offset 16*half, v4..7 at 32+16*half within 64B k-chunk
// ---------------------------------------------------------------------------
__global__ __launch_bounds__(256) void gemm_wmma_kernel(
    const unsigned short* __restrict__ xhi, const unsigned short* __restrict__ xlo,
    const unsigned int* __restrict__ bhi, const unsigned int* __restrict__ blo,
    float* __restrict__ h, int N) {
    int wave = blockIdx.x * (blockDim.x >> 5) + (threadIdx.x >> 5);
    int lane = threadIdx.x & 31;
    int ngrp  = wave & 7;   // 8 groups of 4 n-tiles (HC = 512 = 8*4*16)
    int mtile = wave >> 3;
    if (mtile * 16 >= N) return;  // wave-uniform guard (EXEC stays all-ones)
    int m = lane & 15;
    int halfA = lane >> 4;
    // per-lane A base: row start + 8*half shorts (16*half bytes)
    const unsigned short* xa = xhi + (size_t)(mtile * 16 + m) * 512 + 8 * halfA;
    const unsigned short* xb = xlo + (size_t)(mtile * 16 + m) * 512 + 8 * halfA;
    // per-lane B bases (dwords); k-step stride = 256 dwords
    const unsigned int* bp0 = bhi + ((size_t)(ngrp * 4 + 0) * 512 + lane) * 8;
    const unsigned int* bp1 = bhi + ((size_t)(ngrp * 4 + 1) * 512 + lane) * 8;
    const unsigned int* bp2 = bhi + ((size_t)(ngrp * 4 + 2) * 512 + lane) * 8;
    const unsigned int* bp3 = bhi + ((size_t)(ngrp * 4 + 3) * 512 + lane) * 8;
    const unsigned int* lp0 = blo + ((size_t)(ngrp * 4 + 0) * 512 + lane) * 8;
    const unsigned int* lp1 = blo + ((size_t)(ngrp * 4 + 1) * 512 + lane) * 8;
    const unsigned int* lp2 = blo + ((size_t)(ngrp * 4 + 2) * 512 + lane) * 8;
    const unsigned int* lp3 = blo + ((size_t)(ngrp * 4 + 3) * 512 + lane) * 8;

    v8f c0 = {0.f,0.f,0.f,0.f,0.f,0.f,0.f,0.f};
    v8f c1 = c0, c2 = c0, c3 = c0;
    for (int ks = 0; ks < 16; ++ks) {
        ABu ah, al, bh0, bl0, bh1, bl1, bh2, bl2, bh3, bl3;
        int ka = ks * 32;       // shorts
        int kb = ks * 256;      // dwords
        ah.q[0] = *(const v4u*)(xa + ka);
        ah.q[1] = *(const v4u*)(xa + ka + 16);
        al.q[0] = *(const v4u*)(xb + ka);
        al.q[1] = *(const v4u*)(xb + ka + 16);
        bh0.q[0] = *(const v4u*)(bp0 + kb); bh0.q[1] = *(const v4u*)(bp0 + kb + 4);
        bh1.q[0] = *(const v4u*)(bp1 + kb); bh1.q[1] = *(const v4u*)(bp1 + kb + 4);
        bh2.q[0] = *(const v4u*)(bp2 + kb); bh2.q[1] = *(const v4u*)(bp2 + kb + 4);
        bh3.q[0] = *(const v4u*)(bp3 + kb); bh3.q[1] = *(const v4u*)(bp3 + kb + 4);
        bl0.q[0] = *(const v4u*)(lp0 + kb); bl0.q[1] = *(const v4u*)(lp0 + kb + 4);
        bl1.q[0] = *(const v4u*)(lp1 + kb); bl1.q[1] = *(const v4u*)(lp1 + kb + 4);
        bl2.q[0] = *(const v4u*)(lp2 + kb); bl2.q[1] = *(const v4u*)(lp2 + kb + 4);
        bl3.q[0] = *(const v4u*)(lp3 + kb); bl3.q[1] = *(const v4u*)(lp3 + kb + 4);
        // interleave the four independent accumulator chains
        c0 = __builtin_amdgcn_wmma_f32_16x16x32_bf16(false, ah.v, false, bh0.v, (short)0, c0, false, false);
        c1 = __builtin_amdgcn_wmma_f32_16x16x32_bf16(false, ah.v, false, bh1.v, (short)0, c1, false, false);
        c2 = __builtin_amdgcn_wmma_f32_16x16x32_bf16(false, ah.v, false, bh2.v, (short)0, c2, false, false);
        c3 = __builtin_amdgcn_wmma_f32_16x16x32_bf16(false, ah.v, false, bh3.v, (short)0, c3, false, false);
        c0 = __builtin_amdgcn_wmma_f32_16x16x32_bf16(false, ah.v, false, bl0.v, (short)0, c0, false, false);
        c1 = __builtin_amdgcn_wmma_f32_16x16x32_bf16(false, ah.v, false, bl1.v, (short)0, c1, false, false);
        c2 = __builtin_amdgcn_wmma_f32_16x16x32_bf16(false, ah.v, false, bl2.v, (short)0, c2, false, false);
        c3 = __builtin_amdgcn_wmma_f32_16x16x32_bf16(false, ah.v, false, bl3.v, (short)0, c3, false, false);
        c0 = __builtin_amdgcn_wmma_f32_16x16x32_bf16(false, al.v, false, bh0.v, (short)0, c0, false, false);
        c1 = __builtin_amdgcn_wmma_f32_16x16x32_bf16(false, al.v, false, bh1.v, (short)0, c1, false, false);
        c2 = __builtin_amdgcn_wmma_f32_16x16x32_bf16(false, al.v, false, bh2.v, (short)0, c2, false, false);
        c3 = __builtin_amdgcn_wmma_f32_16x16x32_bf16(false, al.v, false, bh3.v, (short)0, c3, false, false);
    }
    // C layout: VGPR v -> row m = v + 8*half, col n = lane&15
    int ncol = lane & 15;
    int mbase = mtile * 16 + halfA * 8;
    float* hp = h + (size_t)mbase * 512 + ngrp * 64 + ncol;
#pragma unroll
    for (int v = 0; v < 8; ++v) {
        size_t r = (size_t)v * 512;
        hp[r]       = c0[v];
        hp[r + 16]  = c1[v];
        hp[r + 32]  = c2[v];
        hp[r + 48]  = c3[v];
    }
}

// a_src[n,h] = <h[n,h,:], att_src[h,:]> ; same for dst. One thread per (node,head).
__global__ void att_proj_kernel(const float* __restrict__ h,
                                const float* __restrict__ att_src,
                                const float* __restrict__ att_dst,
                                float* __restrict__ as_, float* __restrict__ ad_, int N) {
    int t = blockIdx.x * blockDim.x + threadIdx.x;
    if (t >= N * 4) return;
    int node = t >> 2, hd = t & 3;
    const float* hp = h + (size_t)node * 512 + hd * 128;
    const float* sv = att_src + hd * 128;
    const float* dv = att_dst + hd * 128;
    float a1 = 0.f, a2 = 0.f;
#pragma unroll 4
    for (int c = 0; c < 128; ++c) { float v = hp[c]; a1 += v * sv[c]; a2 += v * dv[c]; }
    as_[t] = a1; ad_[t] = a2;
}

__global__ void init_branch_kernel(float* __restrict__ outacc, unsigned* __restrict__ emaxkey,
                                   float* __restrict__ denom, float* __restrict__ gsum, int N) {
    int t = blockIdx.x * blockDim.x + threadIdx.x;
    if (t < N * 512) outacc[t] = 0.f;
    if (t < N * 4) { emaxkey[t] = KEY_NEG_INF; denom[t] = 0.f; }
    if (t < 512) gsum[t] = 0.f;
}

// Pass 1: segment max via order-preserving uint atomicMax
__global__ void edge_max_kernel(const int* __restrict__ adj, const float* __restrict__ as_,
                                const float* __restrict__ ad_, unsigned* __restrict__ emaxkey,
                                int E, int EP) {
    int e = blockIdx.x * blockDim.x + threadIdx.x;
    if (e >= EP) return;
    int s, d;
    if (e < E) { s = adj[e]; d = adj[E + e]; } else { s = d = e - E; }
#pragma unroll
    for (int hd = 0; hd < 4; ++hd) {
        float v = lrelu(as_[s * 4 + hd] + ad_[d * 4 + hd], 0.2f);
        atomicMax(&emaxkey[d * 4 + hd], fkey(v));
    }
}

// Pass 2: ex = exp(e - max); denom += ex
__global__ void edge_expsum_kernel(const int* __restrict__ adj, const float* __restrict__ as_,
                                   const float* __restrict__ ad_, const unsigned* __restrict__ emaxkey,
                                   float* __restrict__ ex, float* __restrict__ denom, int E, int EP) {
    int e = blockIdx.x * blockDim.x + threadIdx.x;
    if (e >= EP) return;
    int s, d;
    if (e < E) { s = adj[e]; d = adj[E + e]; } else { s = d = e - E; }
#pragma unroll
    for (int hd = 0; hd < 4; ++hd) {
        float v = lrelu(as_[s * 4 + hd] + ad_[d * 4 + hd], 0.2f);
        float xv = expf(v - funkey(emaxkey[d * 4 + hd]));
        ex[(size_t)e * 4 + hd] = xv;
        atomicAdd(&denom[d * 4 + hd], xv);
    }
}

// Pass 3: out[dst] += h[src] * alpha  — one wave per edge, 512 cols, coalesced
__global__ __launch_bounds__(256) void edge_msg_kernel(
    const int* __restrict__ adj, const float* __restrict__ ex, const float* __restrict__ denom,
    const float* __restrict__ h, float* __restrict__ outacc, int E, int EP) {
    int wave = blockIdx.x * (blockDim.x >> 5) + (threadIdx.x >> 5);
    if (wave >= EP) return;
    int lane = threadIdx.x & 31;
    int s, d;
    if (wave < E) { s = adj[wave]; d = adj[E + wave]; } else { s = d = wave - E; }
    float alpha[4];
#pragma unroll
    for (int hd = 0; hd < 4; ++hd)
        alpha[hd] = ex[(size_t)wave * 4 + hd] / (denom[d * 4 + hd] + 1e-16f);
    const float* hs = h + (size_t)s * 512;
    float* od = outacc + (size_t)d * 512;
#pragma unroll
    for (int i = 0; i < 16; ++i) {
        int c = lane + i * 32;
        atomicAdd(&od[c], hs[c] * alpha[c >> 7]);
    }
}

// LeakyReLU(out + bias) then column partial-sums for the global mean
__global__ void reduce_cols_kernel(const float* __restrict__ outacc, const float* __restrict__ bias,
                                   float* __restrict__ gsum, int N) {
    int t = blockIdx.x * blockDim.x + threadIdx.x;  // 512 cols x 64 groups
    int c = t & 511, grp = t >> 9;
    float b = bias[c];
    float sum = 0.f;
    for (int n = grp; n < N; n += 64) sum += lrelu(outacc[(size_t)n * 512 + c] + b, 0.01f);
    atomicAdd(&gsum[c], sum);
}

// g = mean; gout = leaky(g @ fc1_w + fc1_b)
__global__ void fc_kernel(const float* __restrict__ gsum, const float* __restrict__ fc1w,
                          const float* __restrict__ fc1b, float* __restrict__ gbuf, int N, int br) {
    int j = blockIdx.x * blockDim.x + threadIdx.x;
    if (j >= 512) return;
    float invN = 1.f / (float)N;
    float acc = fc1b[j];
    for (int c = 0; c < 512; ++c) acc += (gsum[c] * invN) * fc1w[c * 512 + j];
    gbuf[br * 512 + j] = lrelu(acc, 0.01f);
}

__global__ void finalize_kernel(const float* __restrict__ gbuf, float* __restrict__ out) {
    int j = blockIdx.x * blockDim.x + threadIdx.x;
    if (j >= 512) return;
    float a = gbuf[j], b = gbuf[512 + j];
    out[j] = a; out[512 + j] = b; out[1024 + j] = a - b;
}

extern "C" void kernel_launch(void* const* d_in, const int* in_sizes, int n_in,
                              void* d_out, int out_size, void* d_ws, size_t ws_size,
                              hipStream_t stream) {
    const float* x       = (const float*)d_in[0];
    const int*   adj     = (const int*)d_in[1];
    const float* wt_x    = (const float*)d_in[2];
    const int*   wt_adj  = (const int*)d_in[3];
    const float* W       = (const float*)d_in[4];
    const float* att_src = (const float*)d_in[5];
    const float* att_dst = (const float*)d_in[6];
    const float* bias    = (const float*)d_in[7];
    const float* fc1w    = (const float*)d_in[8];
    const float* fc1b    = (const float*)d_in[9];
    float* out = (float*)d_out;

    const int N = in_sizes[0] / 512;   // 20000
    const int E = in_sizes[1] / 2;     // 320000
    const int EP = E + N;              // + self loops
    const int NF = N * 512;

    // workspace carving
    char* p = (char*)d_ws;
    auto alloc = [&](size_t bytes) -> char* {
        char* r = p; p += (bytes + 255) & ~(size_t)255; return r;
    };
    unsigned int*   bhi    = (unsigned int*)alloc(32 * 16 * 32 * 8 * 4);
    unsigned int*   blo    = (unsigned int*)alloc(32 * 16 * 32 * 8 * 4);
    unsigned short* xhi    = (unsigned short*)alloc((size_t)NF * 2);
    unsigned short* xlo    = (unsigned short*)alloc((size_t)NF * 2);
    float*          h      = (float*)alloc((size_t)NF * 4);
    float*          outacc = (float*)alloc((size_t)NF * 4);
    float*          as_    = (float*)alloc((size_t)N * 16);
    float*          ad_    = (float*)alloc((size_t)N * 16);
    unsigned*       emaxk  = (unsigned*)alloc((size_t)N * 16);
    float*          denom  = (float*)alloc((size_t)N * 16);
    float*          ex     = (float*)alloc((size_t)EP * 16);
    float*          gsum   = (float*)alloc(2048);
    float*          gbuf   = (float*)alloc(4096);

    const int B = 256;
    pack_w_kernel<<<(32 * 16 * 32 * 8 + B - 1) / B, B, 0, stream>>>(W, bhi, blo);

    const int mtiles = (N + 15) / 16;
    const int gemm_blocks = (mtiles * 8 + 7) / 8;   // 8 waves/block; 8 n-groups per mtile

    for (int br = 0; br < 2; ++br) {
        const float* xb = br ? wt_x : x;
        const int*   ab = br ? wt_adj : adj;

        conv_x_kernel<<<(NF + B - 1) / B, B, 0, stream>>>(xb, xhi, xlo, NF);
        gemm_wmma_kernel<<<gemm_blocks, B, 0, stream>>>(xhi, xlo, bhi, blo, h, N);
        att_proj_kernel<<<(N * 4 + B - 1) / B, B, 0, stream>>>(h, att_src, att_dst, as_, ad_, N);
        init_branch_kernel<<<(NF + B - 1) / B, B, 0, stream>>>(outacc, emaxk, denom, gsum, N);
        edge_max_kernel<<<(EP + B - 1) / B, B, 0, stream>>>(ab, as_, ad_, emaxk, E, EP);
        edge_expsum_kernel<<<(EP + B - 1) / B, B, 0, stream>>>(ab, as_, ad_, emaxk, ex, denom, E, EP);
        edge_msg_kernel<<<(EP + 7) / 8, B, 0, stream>>>(ab, ex, denom, h, outacc, E, EP);
        reduce_cols_kernel<<<(512 * 64) / B, B, 0, stream>>>(outacc, bias, gsum, N);
        fc_kernel<<<2, B, 0, stream>>>(gsum, fc1w, fc1b, gbuf, N, br);
    }
    finalize_kernel<<<2, B, 0, stream>>>(gbuf, out);
}
